// NaturalAttention_81432579932595
// MI455X (gfx1250) — matrix operand order repaired and checked by reading, not compiled, not run on hardware
//
#include <hip/hip_runtime.h>
#include <stdint.h>
#include <math.h>

// ---------------------------------------------------------------- constants
#define BATCH  4
#define SEQ    1024
#define HID    1024
#define NH     16
#define HD     64
#define MAXREL 128
#define RELN   257          // 2*MAXREL+1 distinct relative positions
#define RELP   272          // R matrix stride: 17 WMMA n-tiles of 16
#define RELWROWS 512        // rel-weight rows padded to one 8-wave*64-col strip

typedef __attribute__((ext_vector_type(16))) __bf16 v16bf;
typedef __attribute__((ext_vector_type(8)))  float  v8f;
typedef __attribute__((ext_vector_type(4)))  unsigned int u32x4;
typedef __attribute__((ext_vector_type(8)))  int i32x8;
typedef __attribute__((ext_vector_type(4)))  int i32x4;

union FragB { uint32_t u[8]; v16bf v; };

__device__ __forceinline__ unsigned short f2bf(float f) {
  uint32_t u = __float_as_uint(f);
  uint32_t r = u + 0x7FFFu + ((u >> 16) & 1u);   // round-to-nearest-even
  return (unsigned short)(r >> 16);
}

// ------------------------------------------------- Tensor Data Mover issue
// Load a 16-row x 32-element bf16 tile (row-major, 64B rows) into LDS.
// D# per CDNA5 ISA ch.8: group0 = {count, lds_addr, gaddr_lo, gaddr_hi|type=2},
// group1 = {data_size=2B, tensor_dim0=K, tensor_dim1=rows, tile 32x16, stride}.
__device__ __forceinline__ void tdm_load_tile(const unsigned short* gsrc,
                                              uint32_t lds_off,
                                              int rows_total, int K, int lda)
{
  uint64_t ga = (uint64_t)(uintptr_t)gsrc;
  u32x4 g0;
  g0[0] = 1u;                                        // count = 1 descriptor
  g0[1] = lds_off;                                   // LDS byte address
  g0[2] = (uint32_t)ga;                              // global addr [31:0]
  g0[3] = ((uint32_t)(ga >> 32) & 0x01FFFFFFu) | (2u << 30);  // [56:32]|type=2
  i32x8 g1;
  g1[0] = 0x00010000;                                // data_size = 2 bytes
  g1[1] = (int)(((uint32_t)K & 0xFFFFu) << 16);                    // dim0 lo
  g1[2] = (int)(((uint32_t)K >> 16) |
                (((uint32_t)rows_total & 0xFFFFu) << 16));         // dim0 hi|dim1 lo
  g1[3] = (int)(((uint32_t)rows_total >> 16) | (32u << 16));       // dim1 hi|tile0=32
  g1[4] = 16;                                        // tile_dim1 = 16 rows
  g1[5] = lda;                                       // tensor_dim0_stride
  g1[6] = 0;
  g1[7] = 0;
  i32x4 zz = {};
#if defined(__clang_major__) && (__clang_major__ >= 23)
  i32x8 z8 = {};
  __builtin_amdgcn_tensor_load_to_lds(g0, g1, zz, zz, z8, 0);
#else
  __builtin_amdgcn_tensor_load_to_lds(g0, g1, zz, zz, 0);
#endif
}

// ---------------------------------------------------------------- converts
__global__ void f32_to_bf16_kernel(const float* __restrict__ src,
                                   unsigned short* __restrict__ dst, int n) {
  int i = blockIdx.x * blockDim.x + threadIdx.x;
  if (i < n) dst[i] = f2bf(src[i]);
}

// rel_emb [257,64] -> bf16 padded to [RELWROWS,64] (pad rows zero)
__global__ void relw_bf16_kernel(const float* __restrict__ src,
                                 unsigned short* __restrict__ dst) {
  int i = blockIdx.x * blockDim.x + threadIdx.x;
  if (i >= RELWROWS * HD) return;
  int j = i >> 6;
  dst[i] = (j < RELN) ? f2bf(src[i]) : (unsigned short)0;
}

// ---------------------------------------------------------------- WMMA GEMM
// C[M,N] = (A[M,K](bf16) @ W[N,K]^T(bf16) + bias) * scale
// Block: 8 waves, tile 16(M) x 512(N). A 16x32 k-slab is TDM-loaded into LDS
// (double buffered) and shared by all waves; each wave register-blocks a
// 16x64 strip (4 accumulators, 4 WMMA per A fragment).
// mode 0: fp32 row-major C (ldc)      -- R matrix, final projection
// mode 1: bf16 head layout  [b,h,s,d] -- Q, K
// mode 3: bf16 head layout transposed [b,h,d,s] -- V (for P@V B-fragments)
__global__ void __launch_bounds__(256)
wmma_gemm_kernel(const unsigned short* __restrict__ A, int lda,
                 const unsigned short* __restrict__ W, int ldw,
                 const float* __restrict__ bias,
                 void* __restrict__ C, int ldc,
                 int Mrows, int N, int K, int mode, float scale)
{
  __shared__ unsigned short abuf[2][16 * 32];        // 2 x 1KB double buffer

  const int lane = threadIdx.x & 31;
  const int wave = threadIdx.x >> 5;
  const int half = lane >> 4;
  const int l16  = lane & 15;

  const int m0  = blockIdx.x * 16;
  const int nw0 = (blockIdx.y * 8 + wave) * 64;      // wave's 64-col strip

  const uint32_t abase = (uint32_t)(uintptr_t)(void*)&abuf[0][0];
  const int KT = K >> 5;

  if (wave == 0)
    tdm_load_tile(A + (size_t)m0 * lda, abase, Mrows, K, lda);

  v8f acc[4] = {};
  for (int kt = 0; kt < KT; ++kt) {
    if (wave == 0) __builtin_amdgcn_s_wait_tensorcnt(0);
    __syncthreads();                                  // tile kt ready for all
    if (wave == 0 && kt + 1 < KT)
      tdm_load_tile(A + (size_t)m0 * lda + (kt + 1) * 32,
                    abase + (uint32_t)(((kt + 1) & 1) * 1024), Mrows, K, lda);

    // A fragment from LDS: per-lane row m=l16; the ISA A-layout makes each
    // half's 8 elements contiguous -> two ds_load_b128.
    FragB a;
    {
      const char* ar = (const char*)(abuf[kt & 1] + l16 * 32);
      *(i32x4*)&a.u[0] = *(const i32x4*)(ar + (half << 4));
      *(i32x4*)&a.u[4] = *(const i32x4*)(ar + 32 + (half << 4));
    }

#pragma unroll
    for (int c = 0; c < 4; ++c) {
      FragB b;
      const unsigned short* Wrow =
          W + (size_t)(nw0 + c * 16 + l16) * ldw + kt * 32;
#pragma unroll
      for (int v = 0; v < 8; ++v) {
        const int kB = ((v >> 2) << 4) + (half << 3) + ((v & 3) << 1);  // B frag: K = (v/4)*16+half*8+2(v%4)
        b.u[v] = *(const uint32_t*)(Wrow + kB);
      }
      acc[c] = __builtin_amdgcn_wmma_f32_16x16x32_bf16(false, a.v, false, b.v,
                                                       (short)0, acc[c],
                                                       false, false);
    }
  }

  // epilogue: C frag row m = r + half*8, col n = lane%16 (per 16-col tile)
#pragma unroll
  for (int c = 0; c < 4; ++c) {
    const int gn = nw0 + c * 16 + l16;
    if (gn >= N) continue;
    const float bn = bias ? bias[gn] : 0.0f;
#pragma unroll
    for (int r = 0; r < 8; ++r) {
      const float val = (acc[c][r] + bn) * scale;
      const int gm = m0 + r + (half << 3);
      if (mode == 0) {
        ((float*)C)[(size_t)gm * ldc + gn] = val;
      } else {
        const int b_ = gm >> 10, s_ = gm & (SEQ - 1); // gm = b*SEQ + s
        const int h_ = gn >> 6,  d_ = gn & (HD - 1);  // gn = h*HD + d
        unsigned short* O = (unsigned short*)C;
        if (mode == 1)
          O[(((size_t)(b_ * NH + h_) * SEQ + s_) << 6) + d_] = f2bf(val);
        else  // mode 3: V transposed
          O[(((size_t)(b_ * NH + h_) * HD + d_) << 10) + s_] = f2bf(val);
      }
    }
  }
}

// ---------------------------------------------------------------- attention
// One wave per 16-query tile; flash-style online softmax over 32-key tiles.
// Qb/Kb: bf16 [bh, s, hd]; Vt: bf16 [bh, hd, s]; R: fp32 [bh*S, RELP] (scaled).
__global__ void __launch_bounds__(128)
attn_kernel(const unsigned short* __restrict__ Qb,
            const unsigned short* __restrict__ Kb,
            const unsigned short* __restrict__ Vt,
            const float* __restrict__ R,
            unsigned short* __restrict__ Ob)
{
  __shared__ unsigned short pbuf[4][16 * 32];   // per-wave P tile (bf16)

  const int lane = threadIdx.x & 31;
  const int wave = threadIdx.x >> 5;
  const int half = lane >> 4;
  const int l16  = lane & 15;

  const int tile = blockIdx.x * 4 + wave;       // 4096 q-tiles total
  const int bh   = tile >> 6;                   // 64 tiles per (b,h)
  const int q0   = (tile & 63) << 4;

  // ---- Q fragments (hd=64 -> two k-steps of 32); Q already scaled by 1/sqrt(d)
  FragB qf[2];
  const unsigned short* Qrow = Qb + ((size_t)bh * SEQ + q0 + l16) * HD;
#pragma unroll
  for (int ks = 0; ks < 2; ++ks)
#pragma unroll
    for (int v = 0; v < 8; ++v) {
      const int kA = ((v >> 2) << 4) + (half << 3) + ((v & 3) << 1);
      qf[ks].u[v] = *(const uint32_t*)(Qrow + ks * 32 + kA);
    }

  v8f oacc[4] = {};
  float mrun[8], lrun[8];
#pragma unroll
  for (int r = 0; r < 8; ++r) { mrun[r] = -1e30f; lrun[r] = 0.0f; }

  const float* Rrow = R + ((size_t)bh * SEQ + q0) * RELP;
  unsigned short* pb = pbuf[wave];

  for (int t = 0; t < 32; ++t) {
    const int k0 = t << 5;

    // ---- S(16x32) = Q @ K^T : two 16-key halves x two hd k-steps
    v8f s0 = {}, s1 = {};
#pragma unroll
    for (int nhf = 0; nhf < 2; ++nhf) {
      const unsigned short* Krow =
          Kb + ((size_t)bh * SEQ + k0 + nhf * 16 + l16) * HD;
#pragma unroll
      for (int ks = 0; ks < 2; ++ks) {
        FragB kf;
#pragma unroll
        for (int v = 0; v < 8; ++v) {
          const int kB = ((v >> 2) << 4) + (half << 3) + ((v & 3) << 1);
          kf.u[v] = *(const uint32_t*)(Krow + ks * 32 + kB);
        }
        if (nhf == 0)
          s0 = __builtin_amdgcn_wmma_f32_16x16x32_bf16(false, qf[ks].v, false,
                                                       kf.v, (short)0, s0,
                                                       false, false);
        else
          s1 = __builtin_amdgcn_wmma_f32_16x16x32_bf16(false, qf[ks].v, false,
                                                       kf.v, (short)0, s1,
                                                       false, false);
      }
    }

    // ---- add gathered relative-position bias (R already scaled)
    float p0[8], p1[8], tmax[8];
#pragma unroll
    for (int r = 0; r < 8; ++r) {
      const int m = r + (half << 3);
      const int q = q0 + m;
      const int key0 = k0 + l16;
      const int d0 = min(max(key0 - q, -MAXREL), MAXREL) + MAXREL;
      const int d1 = min(max(key0 + 16 - q, -MAXREL), MAXREL) + MAXREL;
      const float* rp = Rrow + (size_t)m * RELP;
      p0[r] = s0[r] + rp[d0];
      p1[r] = s1[r] + rp[d1];
      tmax[r] = fmaxf(p0[r], p1[r]);
    }

    // ---- online softmax: row reductions across the 16 lanes of each half
#pragma unroll
    for (int r = 0; r < 8; ++r) {
      float mx = tmax[r];
      mx = fmaxf(mx, __shfl_xor(mx, 1, 32));
      mx = fmaxf(mx, __shfl_xor(mx, 2, 32));
      mx = fmaxf(mx, __shfl_xor(mx, 4, 32));
      mx = fmaxf(mx, __shfl_xor(mx, 8, 32));
      const float mnew  = fmaxf(mrun[r], mx);
      const float alpha = __expf(mrun[r] - mnew);
      mrun[r] = mnew;
      p0[r] = __expf(p0[r] - mnew);
      p1[r] = __expf(p1[r] - mnew);
      float rs = p0[r] + p1[r];
      rs += __shfl_xor(rs, 1, 32);
      rs += __shfl_xor(rs, 2, 32);
      rs += __shfl_xor(rs, 4, 32);
      rs += __shfl_xor(rs, 8, 32);
      lrun[r] = lrun[r] * alpha + rs;
      oacc[0][r] *= alpha; oacc[1][r] *= alpha;
      oacc[2][r] *= alpha; oacc[3][r] *= alpha;
    }

    // ---- P: C-layout -> LDS (row-major 16x32 bf16) -> A-fragment
#pragma unroll
    for (int r = 0; r < 8; ++r) {
      const int m = r + (half << 3);
      pb[m * 32 + l16]      = f2bf(p0[r]);
      pb[m * 32 + 16 + l16] = f2bf(p1[r]);
    }
    asm volatile("s_wait_dscnt 0" ::: "memory");
    FragB pf;
    {
      const char* pr = (const char*)(pb + l16 * 32);
      *(i32x4*)&pf.u[0] = *(const i32x4*)(pr + (half << 4));
      *(i32x4*)&pf.u[4] = *(const i32x4*)(pr + 32 + (half << 4));
    }

    // ---- O(16x64) += P(16x32) @ V(32x64), 4 d-chunks of 16
#pragma unroll
    for (int c = 0; c < 4; ++c) {
      FragB vf;
      const unsigned short* Vrow =
          Vt + ((size_t)bh * HD + c * 16 + l16) * SEQ + k0;
#pragma unroll
      for (int v = 0; v < 8; ++v) {
        const int kB = ((v >> 2) << 4) + (half << 3) + ((v & 3) << 1);
        vf.u[v] = *(const uint32_t*)(Vrow + kB);
      }
      oacc[c] = __builtin_amdgcn_wmma_f32_16x16x32_bf16(false, pf.v, false,
                                                        vf.v, (short)0,
                                                        oacc[c], false, false);
    }
  }

  // ---- epilogue: normalize and store bf16 in [b, s, h*HD+d] for out-proj
  const int b_ = bh >> 4, h_ = bh & 15;
#pragma unroll
  for (int r = 0; r < 8; ++r) {
    const float inv = 1.0f / lrun[r];
    const int s_ = q0 + r + (half << 3);
    const size_t base = ((size_t)(b_ * SEQ + s_) * HID) + (size_t)h_ * HD;
#pragma unroll
    for (int c = 0; c < 4; ++c)
      Ob[base + c * 16 + l16] = f2bf(oacc[c][r] * inv);
  }
}

// ---------------------------------------------------------------- launcher
extern "C" void kernel_launch(void* const* d_in, const int* in_sizes, int n_in,
                              void* d_out, int out_size, void* d_ws, size_t ws_size,
                              hipStream_t stream)
{
  const float* x   = (const float*)d_in[0];
  const float* Wq  = (const float*)d_in[1];
  const float* bq  = (const float*)d_in[2];
  const float* Wk  = (const float*)d_in[3];
  const float* bk  = (const float*)d_in[4];
  const float* Wv  = (const float*)d_in[5];
  const float* bv  = (const float*)d_in[6];
  const float* Wo  = (const float*)d_in[7];
  const float* bo  = (const float*)d_in[8];
  const float* rel = (const float*)d_in[9];

  const size_t M   = (size_t)BATCH * SEQ;       // 4096 rows
  const size_t BH  = (size_t)BATCH * NH;        // 64 batch-heads

  char* ws = (char*)d_ws;
  size_t off = 0;
  auto alloc = [&](size_t bytes) -> char* {
    char* p = ws + off;
    off = (off + bytes + 255) & ~(size_t)255;
    return p;
  };
  unsigned short* xb   = (unsigned short*)alloc(M * HID * 2);
  unsigned short* Wqb  = (unsigned short*)alloc((size_t)HID * HID * 2);
  unsigned short* Wkb  = (unsigned short*)alloc((size_t)HID * HID * 2);
  unsigned short* Wvb  = (unsigned short*)alloc((size_t)HID * HID * 2);
  unsigned short* Wob  = (unsigned short*)alloc((size_t)HID * HID * 2);
  unsigned short* relW = (unsigned short*)alloc((size_t)RELWROWS * HD * 2);
  unsigned short* Qb   = (unsigned short*)alloc(BH * SEQ * HD * 2);
  unsigned short* Kb   = (unsigned short*)alloc(BH * SEQ * HD * 2);
  unsigned short* Vt   = (unsigned short*)alloc(BH * SEQ * HD * 2);
  float*          Rm   = (float*)alloc(BH * SEQ * RELP * 4);
  unsigned short* Ob   = (unsigned short*)alloc(M * HID * 2);

  const float inv_sqrt_d = 1.0f / sqrtf((float)HD);

  // 1) bf16 conversions
  {
    int n = (int)(M * HID);
    f32_to_bf16_kernel<<<(n + 255) / 256, 256, 0, stream>>>(x, xb, n);
    n = HID * HID;
    f32_to_bf16_kernel<<<(n + 255) / 256, 256, 0, stream>>>(Wq, Wqb, n);
    f32_to_bf16_kernel<<<(n + 255) / 256, 256, 0, stream>>>(Wk, Wkb, n);
    f32_to_bf16_kernel<<<(n + 255) / 256, 256, 0, stream>>>(Wv, Wvb, n);
    f32_to_bf16_kernel<<<(n + 255) / 256, 256, 0, stream>>>(Wo, Wob, n);
    relw_bf16_kernel<<<(RELWROWS * HD + 255) / 256, 256, 0, stream>>>(rel, relW);
  }

  dim3 blk(256);

  // 2) QKV projections (Q pre-scaled by 1/sqrt(hd)); block = 16M x 512N
  dim3 g1((unsigned)(M / 16), HID / 512);
  wmma_gemm_kernel<<<g1, blk, 0, stream>>>(xb, HID, Wqb, HID, bq, Qb, 0,
                                           (int)M, HID, HID, 1, inv_sqrt_d);
  wmma_gemm_kernel<<<g1, blk, 0, stream>>>(xb, HID, Wkb, HID, bk, Kb, 0,
                                           (int)M, HID, HID, 1, 1.0f);
  wmma_gemm_kernel<<<g1, blk, 0, stream>>>(xb, HID, Wvb, HID, bv, Vt, 0,
                                           (int)M, HID, HID, 3, 1.0f);

  // 3) R = Qscaled @ rel_emb^T  [BH*S, RELP]  (turns einsum into a GEMM)
  dim3 g2((unsigned)(BH * SEQ / 16), 1);
  wmma_gemm_kernel<<<g2, blk, 0, stream>>>(Qb, HD, relW, HD, nullptr, Rm, RELP,
                                           (int)(BH * SEQ), RELP, HD, 0, 1.0f);

  // 4) flash attention with relative bias gather
  attn_kernel<<<(unsigned)(BH * SEQ / 16 / 4), 128, 0, stream>>>(Qb, Kb, Vt,
                                                                 Rm, Ob);

  // 5) output projection -> fp32 d_out
  wmma_gemm_kernel<<<g1, blk, 0, stream>>>(Ob, HID, Wob, HID, bo,
                                           (float*)d_out, HID,
                                           (int)M, HID, HID, 0, 1.0f);
}
